// KGRL_1108101562593
// MI455X (gfx1250) — matrix operand order repaired and checked by reading, not compiled, not run on hardware
//
#include <hip/hip_runtime.h>
#include <hip/hip_bf16.h>

// ---------------------------------------------------------------------------
// KGRL forward for MI455X (gfx1250, wave32, WMMA + TDM).
// All GEMMs via v_wmma_f32_16x16x32_bf16 (f32 accumulate). bf16 A-operands
// are staged to LDS by the Tensor Data Mover (tensor_load_to_lds, double
// buffered, TENSORcnt-pipelined); f32/int operands are converted in-flight
// via v_cvt_pk_bf16_f32 with a register-prefetch software pipeline.
// GEMM: block tile 128x64 (BK=32), 256 thr = 8 waves (4M x 2N), wave = 32x32.
// ---------------------------------------------------------------------------

#define B_ 8192
#define E_ 4096
#define DEMB_ 256
#define HID_ 1024
#define DIS_ 128
#define SYM_ 512

typedef __attribute__((ext_vector_type(16))) __bf16 v16bf;
typedef __attribute__((ext_vector_type(8)))  __bf16 v8bf;
typedef __attribute__((ext_vector_type(2)))  __bf16 v2bf;
typedef __attribute__((ext_vector_type(2)))  float  v2f;
typedef __attribute__((ext_vector_type(8)))  float  v8f;
typedef unsigned int uint32x4 __attribute__((ext_vector_type(4)));
typedef int          int32x8  __attribute__((ext_vector_type(8)));
typedef int          int32x4  __attribute__((ext_vector_type(4)));

enum { AM_F32 = 0, AM_BF16 = 1, AM_INT = 2 };
enum { BM_F32 = 0, BM_BF16 = 1 };
enum { EP_BF16 = 0, EP_RELU_BF16 = 1, EP_SIGF32 = 2, EP_F32 = 3, EP_FINAL = 4 };

// Single v_cvt_pk_bf16_f32: pack two f32 -> one dword of 2 bf16.
static __device__ __forceinline__ unsigned int pk2(float a, float b) {
  v2f f = {a, b};
  v2bf h = __builtin_convertvector(f, v2bf);
  union { v2bf v; unsigned int u; } c; c.v = h; return c.u;
}
static __device__ __forceinline__ unsigned short f2bf(float f) {
  union { __bf16 h; unsigned short u; } c; c.h = (__bf16)f; return c.u;
}
static __device__ __forceinline__ float bf2f(unsigned short h) {
  union { unsigned int u; float f; } v; v.u = ((unsigned int)h) << 16; return v.f;
}
static __device__ __forceinline__ float sigmoidf_(float x) {
  return 1.0f / (1.0f + __expf(-x));
}

// TDM: DMA a [128 x 32]-bf16 tile (row stride = Kdim elements) into LDS.
// D# per CDNA5 ISA ch8: group0 = count|lds_addr|global_addr|type=2,
// group1 = data_size=2B, tensor_dim0=Kdim, tensor_dim1=Mdim,
// tile_dim0=32, tile_dim1=128, tensor_dim0_stride=Kdim. Groups 2/3 = 0 (2D).
static __device__ __forceinline__ void tdm_load_a_tile(
    unsigned int lds_byte_addr, const unsigned short* gsrc,
    unsigned int Kdim, unsigned int Mdim) {
  unsigned long long ga = (unsigned long long)(uintptr_t)gsrc;
  uint32x4 g0;
  g0[0] = 1u;                                                   // count=1 (valid)
  g0[1] = lds_byte_addr;                                        // lds_addr
  g0[2] = (unsigned int)ga;                                     // global_addr[31:0]
  g0[3] = (unsigned int)((ga >> 32) & 0x01FFFFFFull) | (2u << 30);  // ga[56:32] | type=2
  int32x8 g1;
  g1[0] = (int)(1u << 16);                                      // wg_mask=0, data_size=2B
  g1[1] = (int)((Kdim & 0xFFFFu) << 16);                        // tensor_dim0[15:0] @48
  g1[2] = (int)(((Kdim >> 16) & 0xFFFFu) | ((Mdim & 0xFFFFu) << 16));
  g1[3] = (int)(((Mdim >> 16) & 0xFFFFu) | (32u << 16));        // tile_dim0 = 32
  g1[4] = (int)128u;                                            // tile_dim1 = 128
  g1[5] = (int)Kdim;                                            // dim0_stride[31:0]
  g1[6] = 0;                                                    // stride hi / dim1_stride
  g1[7] = 0;
  int32x4 z4 = {0, 0, 0, 0};
  int32x8 z8 = {0, 0, 0, 0, 0, 0, 0, 0};
  // 6-arg amdgpu-toolchain (clang-23) form
  __builtin_amdgcn_tensor_load_to_lds(g0, g1, z4, z4, z8, 0);
}

// ---------------------------------------------------------------------------
// C = EPI(A @ B + bias).  A:[M,K] rm, B:[K,N] rm.
// Requires M%128==0, N%64==0, K%32==0 (true for all call sites).
// ---------------------------------------------------------------------------
template <int AMODE, int BMODE, int EPI, bool BIAS>
__global__ __launch_bounds__(256) void gemm_wmma(
    const void* __restrict__ Ap, const void* __restrict__ Bp,
    const float* __restrict__ bias, void* __restrict__ Cp,
    int M, int N, int K,
    const float* __restrict__ muP, const float* __restrict__ sympP,
    const int* __restrict__ sflagP, const int* __restrict__ smaskP) {
  constexpr bool USE_TDM = (AMODE == AM_BF16);       // bf16 A: DMA via TDM
  constexpr int ATILE = 128 * 32;
  __shared__ __align__(16) unsigned short As[(USE_TDM ? 2 : 1) * ATILE];
  __shared__ __align__(16) unsigned short Bst[64 * 32];  // [n][k] transposed

  const int tid  = threadIdx.x;
  const int wave = tid >> 5;
  const int lane = tid & 31;
  const int wm = wave & 3;        // 4 wave-rows of 32
  const int wn = wave >> 2;       // 2 wave-cols of 32
  const int mblk = blockIdx.y * 128;
  const int nblk = blockIdx.x * 64;

  const int row = lane & 15;      // A-frag row / B-frag col within 16-tile
  const int kh  = lane >> 4;      // lane-half selector

  float4 aF[4]; int4 aI[4];       // prefetch registers (mode-dependent liveness)
  float4 bF[2]; uint4 bH;

  auto fetchA = [&](int kt) {
    if (AMODE == AM_F32) {
#pragma unroll
      for (int j = 0; j < 4; ++j) {
        int idx = j * 256 + tid, m = idx >> 3, k = (idx & 7) << 2;
        aF[j] = *(const float4*)((const float*)Ap + (size_t)(mblk + m) * K + kt + k);
      }
    } else if (AMODE == AM_INT) {
#pragma unroll
      for (int j = 0; j < 4; ++j) {
        int idx = j * 256 + tid, m = idx >> 3, k = (idx & 7) << 2;
        aI[j] = *(const int4*)((const int*)Ap + (size_t)(mblk + m) * K + kt + k);
      }
    }
  };
  auto fetchB = [&](int kt) {
    if (BMODE == BM_F32) {
#pragma unroll
      for (int j = 0; j < 2; ++j) {
        int idx = j * 256 + tid, kk = idx >> 4, n = (idx & 15) << 2;
        bF[j] = *(const float4*)((const float*)Bp + (size_t)(kt + kk) * N + nblk + n);
      }
    } else {
      int kk = tid >> 3, n = (tid & 7) << 3;
      bH = *(const uint4*)((const unsigned short*)Bp + (size_t)(kt + kk) * N + nblk + n);
    }
  };

  auto stageA = [&]() {
    if (AMODE == AM_F32) {
#pragma unroll
      for (int j = 0; j < 4; ++j) {
        int idx = j * 256 + tid, m = idx >> 3, k = (idx & 7) << 2;
        *(uint2*)(As + m * 32 + k) = make_uint2(pk2(aF[j].x, aF[j].y), pk2(aF[j].z, aF[j].w));
      }
    } else if (AMODE == AM_INT) {
#pragma unroll
      for (int j = 0; j < 4; ++j) {
        int idx = j * 256 + tid, m = idx >> 3, k = (idx & 7) << 2;
        int iv[4] = {aI[j].x, aI[j].y, aI[j].z, aI[j].w};
        unsigned int p[2];
#pragma unroll
        for (int h = 0; h < 2; ++h) {
          unsigned int lo = (iv[2 * h]     == 1) ? 0x3F80u : ((iv[2 * h]     == -1) ? 0xBF80u : 0u);
          unsigned int hi = (iv[2 * h + 1] == 1) ? 0x3F80u : ((iv[2 * h + 1] == -1) ? 0xBF80u : 0u);
          p[h] = lo | (hi << 16);
        }
        *(uint2*)(As + m * 32 + k) = make_uint2(p[0], p[1]);
      }
    }
  };
  auto stageB = [&]() {
    if (BMODE == BM_F32) {
#pragma unroll
      for (int j = 0; j < 2; ++j) {
        int idx = j * 256 + tid, kk = idx >> 4, n = (idx & 15) << 2;
        Bst[(n + 0) * 32 + kk] = f2bf(bF[j].x);
        Bst[(n + 1) * 32 + kk] = f2bf(bF[j].y);
        Bst[(n + 2) * 32 + kk] = f2bf(bF[j].z);
        Bst[(n + 3) * 32 + kk] = f2bf(bF[j].w);
      }
    } else {
      int kk = tid >> 3, n = (tid & 7) << 3;
      unsigned int w[4] = {bH.x, bH.y, bH.z, bH.w};
#pragma unroll
      for (int t = 0; t < 4; ++t) {
        Bst[(n + 2 * t + 0) * 32 + kk] = (unsigned short)(w[t] & 0xFFFFu);
        Bst[(n + 2 * t + 1) * 32 + kk] = (unsigned short)(w[t] >> 16);
      }
    }
  };

  v8f acc00 = {0.f,0.f,0.f,0.f,0.f,0.f,0.f,0.f}, acc01 = acc00, acc10 = acc00, acc11 = acc00;

  // ---- prologue: start tile 0 ----
  if (USE_TDM) {
    if (wave == 0)
      tdm_load_a_tile((unsigned int)(uintptr_t)As,
                      (const unsigned short*)Ap + (size_t)mblk * K, K, M);
  } else {
    fetchA(0);
  }
  fetchB(0);

  int buf = 0;
  for (int kt = 0; kt < K; kt += 32) {
    const bool more = (kt + 32 < K);
    __syncthreads();               // prior iteration's LDS reads retired
    stageB();
    if (!USE_TDM) stageA();
    if (USE_TDM && wave == 0) {
      if (more) {                  // issue next DMA into other buffer, then
        tdm_load_a_tile((unsigned int)(uintptr_t)(As + (buf ^ 1) * ATILE),
                        (const unsigned short*)Ap + (size_t)mblk * K + kt + 32, K, M);
        __builtin_amdgcn_s_wait_tensorcnt(1);  // current buffer's DMA done
      } else {
        __builtin_amdgcn_s_wait_tensorcnt(0);
      }
    }
    __syncthreads();               // LDS tiles ready for all waves
    if (more) { fetchB(kt + 32); if (!USE_TDM) fetchA(kt + 32); }

    const unsigned short* Ab = As + (USE_TDM ? buf * ATILE : 0);
    // A 16x32 bf16 frag: elem i -> K = kh*16 + i (contiguous, matches B)
    const unsigned short* ap0 = Ab + ((wm * 32 + row) * 32 + kh * 16);
    const unsigned short* ap1 = ap0 + 16 * 32;
    v8bf x0 = *(const v8bf*)ap0,  x1 = *(const v8bf*)(ap0 + 8);
    v8bf y0 = *(const v8bf*)ap1,  y1 = *(const v8bf*)(ap1 + 8);
    v16bf a0 = __builtin_shufflevector(x0, x1, 0,1,2,3,4,5,6,7,8,9,10,11,12,13,14,15);
    v16bf a1 = __builtin_shufflevector(y0, y1, 0,1,2,3,4,5,6,7,8,9,10,11,12,13,14,15);
    // B 32x16 bf16 frag: elem i -> K = kh*16 + i (contiguous 32B per lane)
    const unsigned short* bp0 = Bst + ((wn * 32 + row) * 32 + kh * 16);
    v16bf b0 = *(const v16bf*)bp0;
    v16bf b1 = *(const v16bf*)(bp0 + 16 * 32);

    acc00 = __builtin_amdgcn_wmma_f32_16x16x32_bf16(false, a0, false, b0, (short)0, acc00, false, false);
    acc01 = __builtin_amdgcn_wmma_f32_16x16x32_bf16(false, a0, false, b1, (short)0, acc01, false, false);
    acc10 = __builtin_amdgcn_wmma_f32_16x16x32_bf16(false, a1, false, b0, (short)0, acc10, false, false);
    acc11 = __builtin_amdgcn_wmma_f32_16x16x32_bf16(false, a1, false, b1, (short)0, acc11, false, false);
    buf ^= 1;
  }

  // C/D layout: lanes 0-15 vgpr r -> M=r ; lanes 16-31 vgpr r -> M=8+r ; N = lane&15
  const int mrow = (lane >> 4) * 8;
  const int ncol = lane & 15;
  auto emit = [&](float v, int m, int n) {
    if (BIAS) v += bias[n];
    size_t o = (size_t)m * N + n;
    if (EPI == EP_RELU_BF16) {
      ((unsigned short*)Cp)[o] = f2bf(fmaxf(v, 0.0f));
    } else if (EPI == EP_BF16) {
      ((unsigned short*)Cp)[o] = f2bf(v);
    } else if (EPI == EP_SIGF32) {
      ((float*)Cp)[o] = sigmoidf_(v);
    } else if (EPI == EP_F32) {
      ((float*)Cp)[o] = v;
    } else {  // EP_FINAL: fused filter_sig + mu blend + masks
      float tf = v > 0.0f ? sigmoidf_(v) : v;
      float p  = muP[m];
      float sp = sympP[o];
      float g  = (float)(sflagP[o] * smaskP[o]);
      ((float*)Cp)[o] = (p * sp + (1.0f - p) * tf) * g;
    }
  };
#pragma unroll
  for (int r = 0; r < 8; ++r) {
    int m0 = mblk + wm * 32 + mrow + r;
    int n0 = nblk + wn * 32 + ncol;
    emit(acc00[r], m0,      n0);
    emit(acc01[r], m0,      n0 + 16);
    emit(acc10[r], m0 + 16, n0);
    emit(acc11[r], m0 + 16, n0 + 16);
  }
}

// ---------------------------------------------------------------------------
// mu = sigmoid(h_m @ Wm2 + bm2) : N=1 GEMV, one wave32 per row.
// ---------------------------------------------------------------------------
__global__ __launch_bounds__(256) void mu_kernel(
    const unsigned short* __restrict__ hm, const float* __restrict__ Wm2,
    const float* __restrict__ bm2, float* __restrict__ mu) {
  int wave = threadIdx.x >> 5, lane = threadIdx.x & 31;
  int r = blockIdx.x * 8 + wave;
  float acc = 0.0f;
#pragma unroll 4
  for (int k = lane; k < HID_; k += 32)
    acc += bf2f(hm[(size_t)r * HID_ + k]) * Wm2[k];
#pragma unroll
  for (int off = 16; off > 0; off >>= 1) acc += __shfl_down(acc, off, 32);
  if (lane == 0) mu[r] = sigmoidf_(acc + bm2[0]);
}

// ---------------------------------------------------------------------------
// disease_dis = softmax(logits, axis=-1) * disease_mask : one wave per row.
// ---------------------------------------------------------------------------
__global__ __launch_bounds__(256) void softmax_mask_kernel(
    const float* __restrict__ logits, const int* __restrict__ dmask,
    float* __restrict__ out) {
  int wave = threadIdx.x >> 5, lane = threadIdx.x & 31;
  int r = blockIdx.x * 8 + wave;
  float v[4];
#pragma unroll
  for (int j = 0; j < 4; ++j) v[j] = logits[(size_t)r * DIS_ + lane + j * 32];
  float mx = fmaxf(fmaxf(v[0], v[1]), fmaxf(v[2], v[3]));
#pragma unroll
  for (int off = 16; off > 0; off >>= 1) mx = fmaxf(mx, __shfl_xor(mx, off, 32));
  float e[4], s = 0.0f;
#pragma unroll
  for (int j = 0; j < 4; ++j) { e[j] = __expf(v[j] - mx); s += e[j]; }
#pragma unroll
  for (int off = 16; off > 0; off >>= 1) s += __shfl_xor(s, off, 32);
  float inv = 1.0f / s;
#pragma unroll
  for (int j = 0; j < 4; ++j) {
    int c = lane + j * 32;
    out[(size_t)r * DIS_ + c] = e[j] * inv * (float)dmask[(size_t)r * DIS_ + c];
  }
}

// ---------------------------------------------------------------------------
extern "C" void kernel_launch(void* const* d_in, const int* in_sizes, int n_in,
                              void* d_out, int out_size, void* d_ws, size_t ws_size,
                              hipStream_t stream) {
  const int*   dialog_state  = (const int*)d_in[0];     // [B, DIS+SYM]
  const float* dialog_symrep = (const float*)d_in[1];   // [B, 1, E] == [B, E]
  const float* kg_adj        = (const float*)d_in[2];   // [E, E]
  const int*   sym_flag      = (const int*)d_in[3];     // [B, SYM]
  const int*   sym_mask      = (const int*)d_in[4];     // [B, SYM]
  const int*   disease_mask  = (const int*)d_in[5];     // [B, DIS]
  const int*   symptoms_mask = (const int*)d_in[6];     // [B, SYM]
  const float* W_g  = (const float*)d_in[7];            // [E, DEMB]
  const float* Wd1  = (const float*)d_in[8];            // [SYM, DIS]
  const float* bd1  = (const float*)d_in[9];
  const float* W1   = (const float*)d_in[10];           // [DEMB, HID]
  const float* b1   = (const float*)d_in[11];
  const float* W2   = (const float*)d_in[12];           // [HID, SYM]
  const float* b2   = (const float*)d_in[13];
  const float* Wm1  = (const float*)d_in[14];           // [DIS+SYM, HID]
  const float* bm1  = (const float*)d_in[15];
  const float* Wm2  = (const float*)d_in[16];           // [HID, 1]
  const float* bm2  = (const float*)d_in[17];
  const float* m_d  = (const float*)d_in[18];           // [DIS, SYM]

  float* out_disease = (float*)d_out;                      // [B, DIS]
  float* out_symptom = (float*)d_out + (size_t)B_ * DIS_;  // [B, SYM]

  // Workspace layout (~42 MB total)
  char* ws = (char*)d_ws;
  unsigned short* g_graph  = (unsigned short*)(ws);                     //  2 MB [E,DEMB] bf16
  unsigned short* g_state  = (unsigned short*)(ws + (2u << 20));        //  4 MB [B,DEMB] bf16
  unsigned short* g_big    = (unsigned short*)(ws + (6u << 20));        // 16 MB [B,HID] bf16 (hidden, then h_m)
  float*          g_symp   = (float*)(ws + (22u << 20));                // 16 MB [B,SYM] f32
  float*          g_mu     = (float*)(ws + (38u << 20));                // 32 KB [B]
  float*          g_logits = (float*)(ws + (38u << 20) + (64u << 10));  //  4 MB [B,DIS]

  dim3 blk(256);

  // 1) graph_rep = relu(kg_adj @ W_g)              [4096 x 256], K=4096
  gemm_wmma<AM_F32, BM_F32, EP_RELU_BF16, false>
      <<<dim3(DEMB_ / 64, E_ / 128), blk, 0, stream>>>(
          kg_adj, W_g, nullptr, g_graph, E_, DEMB_, E_,
          nullptr, nullptr, nullptr, nullptr);

  // 2) state = dialog_sym_rep @ graph_rep          [8192 x 256], K=4096
  gemm_wmma<AM_F32, BM_BF16, EP_BF16, false>
      <<<dim3(DEMB_ / 64, B_ / 128), blk, 0, stream>>>(
          dialog_symrep, g_graph, nullptr, g_state, B_, DEMB_, E_,
          nullptr, nullptr, nullptr, nullptr);

  // 3) hidden = relu(state @ W1 + b1)              [8192 x 1024], K=256  (TDM A)
  gemm_wmma<AM_BF16, BM_F32, EP_RELU_BF16, true>
      <<<dim3(HID_ / 64, B_ / 128), blk, 0, stream>>>(
          g_state, W1, b1, g_big, B_, HID_, DEMB_,
          nullptr, nullptr, nullptr, nullptr);

  // 4) symptom_p = sigmoid(hidden @ W2 + b2)       [8192 x 512], K=1024  (TDM A)
  gemm_wmma<AM_BF16, BM_F32, EP_SIGF32, true>
      <<<dim3(SYM_ / 64, B_ / 128), blk, 0, stream>>>(
          g_big, W2, b2, g_symp, B_, SYM_, HID_,
          nullptr, nullptr, nullptr, nullptr);

  // 5) h_m = relu(state_ @ Wm1 + bm1)              [8192 x 1024], K=640
  //    state_ built in-loader from dialog_state ints (==1 -> +1, ==-1 -> -1, else 0)
  gemm_wmma<AM_INT, BM_F32, EP_RELU_BF16, true>
      <<<dim3(HID_ / 64, B_ / 128), blk, 0, stream>>>(
          dialog_state, Wm1, bm1, g_big, B_, HID_, DIS_ + SYM_,
          nullptr, nullptr, nullptr, nullptr);

  // 6) mu = sigmoid(h_m @ Wm2 + bm2)               [8192 x 1] GEMV
  mu_kernel<<<dim3(B_ / 8), blk, 0, stream>>>(g_big, Wm2, bm2, g_mu);

  // 7) logits = sym_mask_f @ Wd1 + bd1             [8192 x 128], K=512
  gemm_wmma<AM_INT, BM_F32, EP_F32, true>
      <<<dim3(DIS_ / 64, B_ / 128), blk, 0, stream>>>(
          sym_mask, Wd1, bd1, g_logits, B_, DIS_, SYM_,
          nullptr, nullptr, nullptr, nullptr);

  // 8) disease_dis = softmax(logits) * disease_mask  -> output 1
  softmax_mask_kernel<<<dim3(B_ / 8), blk, 0, stream>>>(g_logits, disease_mask,
                                                        out_disease);

  // 9) fused: t = disease_dis @ m_d; tf = filter_sig(t);
  //    out2 = (mu*symptom_p + (1-mu)*tf) * sym_flag * symptoms_mask
  gemm_wmma<AM_F32, BM_F32, EP_FINAL, false>
      <<<dim3(SYM_ / 64, B_ / 128), blk, 0, stream>>>(
          out_disease, m_d, nullptr, out_symptom, B_, SYM_, DIS_,
          g_mu, g_symp, sym_flag, symptoms_mask);
}
